// MacridVAE_48515950576398
// MI455X (gfx1250) — matrix operand order repaired and checked by reading, not compiled, not run on hardware
//
#include <hip/hip_runtime.h>
#include <hip/hip_bf16.h>
#include <stdint.h>

typedef __bf16 bf16;
typedef __attribute__((ext_vector_type(16))) __bf16 v16bf;
typedef __attribute__((ext_vector_type(8)))  __bf16 v8bf;
typedef __attribute__((ext_vector_type(8)))  float  v8f;

#define NITEMS 50000
#define BB 256
#define KK 8
#define DD 128
#define HH 600
#define HP 608            // H padded to multiple of 16
#define LD 50016          // N padded to multiple of 32 (contraction leading dim)
#define M2 2048           // B*K
#define KSTEPS 1563       // LD/32
#define NTILES 3125       // NITEMS/16

// ---- WMMA fragment helpers (bf16 16x16x32, contiguous k = 16*(lane/16)+s order, same for A and B) ----
// A (16x32, 16-bit): lane L -> row M=L&15; lanes<16 hold K[0..15],
// lanes>=16 hold K[16..31]  -> two 8-elt chunks at +sel and +sel+8.
static __device__ __forceinline__ v16bf load_a_frag(const bf16* rowk, int lane) {
  int sel = (lane & 16) ? 16 : 0;
  union { v16bf v; v8bf h[2]; } u;
  u.h[0] = *(const v8bf*)(rowk + sel);
  u.h[1] = *(const v8bf*)(rowk + sel + 8);
  return u.v;
}
// B (32x16, 16-bit): lane L -> column n=L&15; lanes<16 hold K[0..15],
// lanes>=16 hold K[16..31] -> one contiguous 16-elt chunk.
static __device__ __forceinline__ v16bf load_b_frag(const bf16* rowk, int lane) {
  return *(const v16bf*)(rowk + ((lane & 16) ? 16 : 0));
}
static __device__ __forceinline__ v8f wmma_bf16(v16bf a, v16bf b, v8f c) {
  return __builtin_amdgcn_wmma_f32_16x16x32_bf16(false, a, false, b, (short)0, c,
                                                 false, false);
}

// ------------------------------- utility ------------------------------------
__global__ void k_zero_u32(uint32_t* p, size_t n) {
  size_t i = (size_t)blockIdx.x * blockDim.x + threadIdx.x;
  size_t st = (size_t)gridDim.x * blockDim.x;
  for (; i < n; i += st) p[i] = 0u;
}

// L2-normalize item rows; emit bf16 row-major [N][D] and transposed [D][LD].
__global__ void k_norm_items(const float* __restrict__ item_emb,
                             bf16* __restrict__ items_bf,
                             bf16* __restrict__ itemsT) {
  int lane = threadIdx.x & 31, wave = threadIdx.x >> 5;
  int n = blockIdx.x * 8 + wave;
  if (n >= NITEMS) return;
  const float* row = item_emb + (size_t)n * DD;
  float v0 = row[lane], v1 = row[lane + 32], v2 = row[lane + 64], v3 = row[lane + 96];
  float s = v0 * v0 + v1 * v1 + v2 * v2 + v3 * v3;
  for (int m = 16; m >= 1; m >>= 1) s += __shfl_xor(s, m, 32);
  float inv = 1.f / fmaxf(sqrtf(s), 1e-12f);
  bf16 b0 = (bf16)(v0 * inv), b1 = (bf16)(v1 * inv);
  bf16 b2 = (bf16)(v2 * inv), b3 = (bf16)(v3 * inv);
  bf16* orow = items_bf + (size_t)n * DD;
  orow[lane] = b0; orow[lane + 32] = b1; orow[lane + 64] = b2; orow[lane + 96] = b3;
  itemsT[(size_t)(lane      ) * LD + n] = b0;
  itemsT[(size_t)(lane + 32 ) * LD + n] = b1;
  itemsT[(size_t)(lane + 64 ) * LD + n] = b2;
  itemsT[(size_t)(lane + 96 ) * LD + n] = b3;
}

__global__ void k_norm_protos(const float* __restrict__ proto, bf16* __restrict__ out) {
  int lane = threadIdx.x & 31, k = threadIdx.x >> 5;  // 8 waves, one per prototype
  const float* row = proto + (size_t)k * DD;
  float v0 = row[lane], v1 = row[lane + 32], v2 = row[lane + 64], v3 = row[lane + 96];
  float s = v0 * v0 + v1 * v1 + v2 * v2 + v3 * v3;
  for (int m = 16; m >= 1; m >>= 1) s += __shfl_xor(s, m, 32);
  float inv = 1.f / fmaxf(sqrtf(s), 1e-12f);
  bf16* o = out + (size_t)k * DD;
  o[lane] = (bf16)(v0 * inv); o[lane + 32] = (bf16)(v1 * inv);
  o[lane + 64] = (bf16)(v2 * inv); o[lane + 96] = (bf16)(v3 * inv);
}

// Transpose W1 [N][H] f32 -> W1T [HP][LD] bf16 (zero-padded), via LDS tiles.
__global__ void k_w1t(const float* __restrict__ W1, bf16* __restrict__ w1t) {
  __shared__ float t[32][33];
  int tx = threadIdx.x & 31, ty = threadIdx.x >> 5;
  int n0 = blockIdx.x * 32, h0 = blockIdx.y * 32;
  for (int i = ty; i < 32; i += 8) {
    int n = n0 + i, h = h0 + tx;
    t[i][tx] = (n < NITEMS && h < HH) ? W1[(size_t)n * HH + h] : 0.f;
  }
  __syncthreads();
  for (int i = ty; i < 32; i += 8) {
    int h = h0 + i, n = n0 + tx;
    if (h < HP && n < LD) w1t[(size_t)h * LD + n] = (bf16)t[tx][i];
  }
}

// ---- cates pass: WMMA logits -> lane-local softmax over K -> xk, colnorms ----
// M-tile of 16 rows = (b0,k0..7),(b1,k0..7). C layout puts the 8 k's of one
// (b,n) into the 8 accumulator VGPRs of a single lane.
__global__ void k_cates(const bf16* __restrict__ xlat, int bstride,
                        const bf16* __restrict__ items_bf,
                        const float* __restrict__ rating,
                        bf16* __restrict__ xk, float* __restrict__ colsq) {
  const int lane = threadIdx.x & 31, wave = threadIdx.x >> 5;
  const int bpair = blockIdx.x;
  v16bf afr[4];
  {
    int m = lane & 15;  // A-frag: lane holds row M=m
    const bf16* arow = xlat + (size_t)(bpair * 2 + (m >> 3)) * bstride + (size_t)(m & 7) * DD;
    for (int kk = 0; kk < 4; kk++) afr[kk] = load_a_frag(arow + kk * 32, lane);
  }
  const int b_lane = bpair * 2 + (lane >> 4);
  const int ncol = lane & 15;
  float sq[KK];
  for (int r = 0; r < KK; r++) sq[r] = 0.f;
  int t = blockIdx.y * 8 + wave;
  const int tstride = gridDim.y * 8;
  for (; t < NTILES; t += tstride) {
    int n_lane = t * 16 + ncol;
    const bf16* brow = items_bf + (size_t)n_lane * DD;
    v8f acc = {};
    for (int kk = 0; kk < 4; kk++)
      acc = wmma_bf16(afr[kk], load_b_frag(brow + kk * 32, lane), acc);
    float l[KK], mx = -1e30f;                      // logits = acc / TAU
    for (int r = 0; r < KK; r++) { l[r] = acc[r] * 10.f; mx = fmaxf(mx, l[r]); }
    float s = 0.f;
    for (int r = 0; r < KK; r++) { l[r] = __expf(l[r] - mx); s += l[r]; }
    float invs = 1.f / s;
    float rt = rating[(size_t)b_lane * NITEMS + n_lane];
    for (int r = 0; r < KK; r++) {
      float x = rt * l[r] * invs;                  // rating * cates
      xk[(size_t)(b_lane * KK + r) * LD + n_lane] = (bf16)x;
      sq[r] += x * x;
    }
  }
  for (int r = 0; r < KK; r++) {                   // reduce within 16-lane half
    float s = sq[r];
    for (int m = 8; m >= 1; m >>= 1) s += __shfl_xor(s, m, 32);
    if (ncol == 0) atomicAdd(&colsq[b_lane * KK + r], s);
  }
}

// ---- split-K GEMM: C[M][ncols] += A[M][LD] * Bt[ncols][LD]ᵀ (bf16->f32) ----
// Each wave computes a 16(M) x 128(N) strip: one A fragment per k-step feeds
// up to 8 accumulators, cutting A (the 205 MB streaming operand) traffic 8x.
__global__ void k_gemm_splitk(const bf16* __restrict__ A, const bf16* __restrict__ Bt,
                              float* __restrict__ C, int ncols, int ntiles,
                              int kchunk) {
  const int lane = threadIdx.x & 31;
  const int t0 = blockIdx.y * 8;
  const int nt = min(8, ntiles - t0);              // wave-uniform
  const bf16* arow = A + (size_t)(blockIdx.x * 16 + (lane & 15)) * LD;
  size_t boff[8];
#pragma unroll
  for (int t = 0; t < 8; t++) {
    int tt = (t < nt) ? t : 0;
    boff[t] = (size_t)((t0 + tt) * 16 + (lane & 15)) * LD + ((lane & 16) ? 16 : 0);
  }
  v8f acc[8];
#pragma unroll
  for (int t = 0; t < 8; t++) acc[t] = {};
  int k0 = blockIdx.z * kchunk;
  int k1 = k0 + kchunk; if (k1 > KSTEPS) k1 = KSTEPS;
  for (int ks = k0; ks < k1; ks++) {
    int kb = ks * 32;
    __builtin_prefetch(arow + kb + 256, 0, 1);
    v16bf af = load_a_frag(arow + kb, lane);
#pragma unroll
    for (int t = 0; t < 8; t++)
      if (t < nt)                                  // uniform branch: EXEC intact
        acc[t] = wmma_bf16(af, *(const v16bf*)(Bt + boff[t] + kb), acc[t]);
  }
  const int rbase = blockIdx.x * 16 + ((lane >> 4) << 3);
#pragma unroll
  for (int t = 0; t < 8; t++) {
    if (t < nt) {
      int col = (t0 + t) * 16 + (lane & 15);
#pragma unroll
      for (int r = 0; r < 8; r++)
        atomicAdd(&C[(size_t)(rbase + r) * ncols + col], acc[t][r]);
    }
  }
}

// l2norm of accumulated latent rows -> next x_latent (column norm cancels here).
__global__ void k_xlat_finish(const float* __restrict__ acc, bf16* __restrict__ xlat_out) {
  int lane = threadIdx.x & 31, wave = threadIdx.x >> 5;
  int row = blockIdx.x * 8 + wave;
  const float* a = acc + (size_t)row * DD;
  float v0 = a[lane], v1 = a[lane + 32], v2 = a[lane + 64], v3 = a[lane + 96];
  float s = v0 * v0 + v1 * v1 + v2 * v2 + v3 * v3;
  for (int m = 16; m >= 1; m >>= 1) s += __shfl_xor(s, m, 32);
  float inv = 1.f / fmaxf(sqrtf(s), 1e-12f);
  bf16* o = xlat_out + (size_t)row * DD;
  o[lane] = (bf16)(v0 * inv); o[lane + 32] = (bf16)(v1 * inv);
  o[lane + 64] = (bf16)(v2 * inv); o[lane + 96] = (bf16)(v3 * inv);
}

// tanh(hacc/||col|| + b1) @ W2 + b2 -> z = l2norm(first 128)
__global__ void k_mlp(const float* __restrict__ hacc, const float* __restrict__ colsq,
                      const float* __restrict__ b1, const float* __restrict__ W2,
                      const float* __restrict__ b2, bf16* __restrict__ z) {
  __shared__ float hs[HH];
  __shared__ float os[256];
  __shared__ float red[128];
  int row = blockIdx.x, tid = threadIdx.x;
  float inv = 1.f / fmaxf(sqrtf(colsq[row]), 1e-12f);
  for (int j = tid; j < HH; j += 256)
    hs[j] = tanhf(hacc[(size_t)row * HP + j] * inv + b1[j]);
  __syncthreads();
  float o = b2[tid];
  for (int j = 0; j < HH; j++) o += hs[j] * W2[(size_t)j * 256 + tid];
  os[tid] = o;
  __syncthreads();
  if (tid < 128) red[tid] = os[tid] * os[tid];
  __syncthreads();
  for (int m = 64; m >= 1; m >>= 1) { if (tid < m) red[tid] += red[tid + m]; __syncthreads(); }
  float invn = 1.f / fmaxf(sqrtf(red[0]), 1e-12f);
  if (tid < 128) z[(size_t)row * DD + tid] = (bf16)(os[tid] * invn);
}

// decode: recompute cates (WMMA) + z·items (WMMA); mixture + log fused lane-local.
__global__ void k_decode(const bf16* __restrict__ xlat, const bf16* __restrict__ zb,
                         const bf16* __restrict__ items_bf,
                         const float* __restrict__ labels,
                         float* __restrict__ out_logits, float* __restrict__ bacc) {
  const int lane = threadIdx.x & 31, wave = threadIdx.x >> 5;
  const int bpair = blockIdx.x;
  v16bf afrC[4], afrZ[4];
  {
    int m = lane & 15;
    size_t off = (size_t)(bpair * 2 + (m >> 3)) * (KK * DD) + (size_t)(m & 7) * DD;
    for (int kk = 0; kk < 4; kk++) {
      afrC[kk] = load_a_frag(xlat + off + kk * 32, lane);
      afrZ[kk] = load_a_frag(zb + off + kk * 32, lane);
    }
  }
  const int b_lane = bpair * 2 + (lane >> 4);
  const int ncol = lane & 15;
  float pS = 0.f, dS = 0.f, lS = 0.f;
  int t = blockIdx.y * 8 + wave;
  const int tstride = gridDim.y * 8;
  for (; t < NTILES; t += tstride) {
    int n_lane = t * 16 + ncol;
    const bf16* brow = items_bf + (size_t)n_lane * DD;
    v8f accC = {}, accZ = {};
    for (int kk = 0; kk < 4; kk++) {
      v16bf bfrag = load_b_frag(brow + kk * 32, lane);
      accC = wmma_bf16(afrC[kk], bfrag, accC);
      accZ = wmma_bf16(afrZ[kk], bfrag, accZ);
    }
    float l[KK], mx = -1e30f;
    for (int r = 0; r < KK; r++) { l[r] = accC[r] * 10.f; mx = fmaxf(mx, l[r]); }
    float s = 0.f;
    for (int r = 0; r < KK; r++) { l[r] = __expf(l[r] - mx); s += l[r]; }
    float num = 0.f;                               // sum_k exp(z·item/tau_dec)*cate_k
    for (int r = 0; r < KK; r++) num += __expf(accZ[r] * 10.f) * l[r];
    float p = num / s;
    float lg = __logf(p);
    out_logits[(size_t)b_lane * NITEMS + n_lane] = lg;
    float lb = labels[(size_t)b_lane * NITEMS + n_lane];
    pS += p; dS += lg * lb; lS += lb;
  }
  for (int m = 8; m >= 1; m >>= 1) {
    pS += __shfl_xor(pS, m, 32); dS += __shfl_xor(dS, m, 32); lS += __shfl_xor(lS, m, 32);
  }
  if (ncol == 0) {
    atomicAdd(&bacc[b_lane * 3 + 0], pS);
    atomicAdd(&bacc[b_lane * 3 + 1], dS);
    atomicAdd(&bacc[b_lane * 3 + 2], lS);
  }
}

// ce = -1/B sum_b (dotL_b - log(sumP_b)*sumL_b)   (ANNEAL=0 -> loss = ce)
__global__ void k_loss(const float* __restrict__ bacc, float* __restrict__ out) {
  __shared__ float red[256];
  int tid = threadIdx.x;
  red[tid] = bacc[tid * 3 + 1] - __logf(bacc[tid * 3 + 0]) * bacc[tid * 3 + 2];
  __syncthreads();
  for (int m = 128; m >= 1; m >>= 1) { if (tid < m) red[tid] += red[tid + m]; __syncthreads(); }
  if (tid == 0) out[0] = -red[0] / (float)BB;
}

extern "C" void kernel_launch(void* const* d_in, const int* in_sizes, int n_in,
                              void* d_out, int out_size, void* d_ws, size_t ws_size,
                              hipStream_t stream) {
  const float* rating    = (const float*)d_in[0];
  const float* labels    = (const float*)d_in[1];
  const float* item_emb  = (const float*)d_in[2];
  const float* proto_emb = (const float*)d_in[3];
  const float* W1        = (const float*)d_in[4];
  const float* b1        = (const float*)d_in[5];
  const float* W2        = (const float*)d_in[6];
  const float* b2        = (const float*)d_in[7];
  float* out_logits = (float*)d_out;
  float* loss_out = out_logits + (size_t)BB * NITEMS;
  (void)in_sizes; (void)n_in; (void)out_size; (void)ws_size;

  uint8_t* ws = (uint8_t*)d_ws;
  size_t off = 0;
  auto carve = [&](size_t bytes) -> uint8_t* {
    uint8_t* p = ws + off; off += (bytes + 255) & ~(size_t)255; return p;
  };
  bf16*  items_bf = (bf16*)carve((size_t)NITEMS * DD * 2); // [N][D]
  bf16*  itemsT   = (bf16*)carve((size_t)DD * LD * 2);     // [D][LD]
  bf16*  w1t      = (bf16*)carve((size_t)HP * LD * 2);     // [HP][LD]
  bf16*  xk       = (bf16*)carve((size_t)M2 * LD * 2);     // [B*K][LD]
  bf16*  xlat0    = (bf16*)carve((size_t)KK * DD * 2);
  bf16*  xlatA    = (bf16*)carve((size_t)M2 * DD * 2);
  bf16*  xlatB    = (bf16*)carve((size_t)M2 * DD * 2);
  bf16*  zb       = (bf16*)carve((size_t)M2 * DD * 2);
  float* xlat_acc = (float*)carve((size_t)M2 * DD * 4);
  float* hacc     = (float*)carve((size_t)M2 * HP * 4);
  float* colsq    = (float*)carve((size_t)M2 * 4);
  float* bacc     = (float*)carve((size_t)BB * 3 * 4);

  auto zero = [&](void* p, size_t bytes) {
    size_t words = bytes / 4;
    size_t nb = (words + 255) / 256; if (nb > 2048) nb = 2048;
    k_zero_u32<<<(int)nb, 256, 0, stream>>>((uint32_t*)p, words);
  };
  zero(itemsT, (size_t)DD * LD * 2);   // tails of padded contraction must be 0
  zero(xk,     (size_t)M2 * LD * 2);
  zero(hacc,   (size_t)M2 * HP * 4);
  zero(bacc,   (size_t)BB * 3 * 4);

  k_norm_items<<<(NITEMS + 7) / 8, 256, 0, stream>>>(item_emb, items_bf, itemsT);
  k_norm_protos<<<1, 256, 0, stream>>>(proto_emb, xlat0);
  dim3 gw1((LD + 31) / 32, (HP + 31) / 32);
  k_w1t<<<gw1, 256, 0, stream>>>(W1, w1t);

  const bf16* xcur = xlat0;
  int bstride = 0;                      // iter 0: broadcast prototypes over batch
  bf16* bufs[3] = { xlatA, xlatB, xlatA };
  for (int it = 0; it < 4; ++it) {
    zero(colsq, (size_t)M2 * 4);
    dim3 gc(BB / 2, 16);
    k_cates<<<gc, 256, 0, stream>>>(xcur, bstride, items_bf, rating, xk, colsq);
    if (it < 3) {
      zero(xlat_acc, (size_t)M2 * DD * 4);
      const int S = 64, kchunk = (KSTEPS + S - 1) / S;
      const int ntiles = DD / 16;                       // 8 -> one N-chunk
      dim3 gg(M2 / 16, 1, S);
      k_gemm_splitk<<<gg, 32, 0, stream>>>(xk, itemsT, xlat_acc, DD, ntiles, kchunk);
      k_xlat_finish<<<M2 / 8, 256, 0, stream>>>(xlat_acc, bufs[it]);
      xcur = bufs[it];
      bstride = KK * DD;
    } else {
      const int S = 16, kchunk = (KSTEPS + S - 1) / S;
      const int ntiles = HP / 16;                       // 38 -> 5 N-chunks
      dim3 gg(M2 / 16, (ntiles + 7) / 8, S);
      k_gemm_splitk<<<gg, 32, 0, stream>>>(xk, w1t, hacc, HP, ntiles, kchunk);
      k_mlp<<<M2, 256, 0, stream>>>(hacc, colsq, b1, W2, b2, zb);
    }
  }
  dim3 gd(BB / 2, 16);
  k_decode<<<gd, 256, 0, stream>>>(xcur, zb, items_bf, labels, out_logits, bacc);
  k_loss<<<1, 256, 0, stream>>>(bacc, loss_out);
}
